// SoftTopKTokenSelect_70944269795791
// MI455X (gfx1250) — matrix operand order repaired and checked by reading, not compiled, not run on hardware
//
#include <hip/hip_runtime.h>
#include <hip/hip_bf16.h>
#include <math.h>

typedef __attribute__((ext_vector_type(16))) _Float16 v16h;
typedef __attribute__((ext_vector_type(8)))  float    v8f;

#define B_  16
#define T_  8192
#define C_  512
#define H_  128   // C/4 hidden
#define N_  128   // regions

// ---------------------------------------------------------------------------
// Kernel 0: prepack W1 [512,128] fp32 -> f16 WMMA B-fragments.
// Fragment f = nt*16 + kc (nt: 16-col hidden tile, kc: 32-row K chunk).
// Element (lane, j) of fragment f maps to  k = kc*32 + (lane<16?0:8) + (j&7) + (j>=8?16:0),
//                                          n = nt*16 + (lane&15)
// (same K order as the A fragment: lanes 0-15 hold K=0..7,16..23,
//  lanes 16-31 hold K=8..15,24..31, packed 2 halves per VGPR).
// ---------------------------------------------------------------------------
__global__ void prepack_w1_kernel(const float* __restrict__ W1,
                                  _Float16* __restrict__ w1p) {
    int gid  = blockIdx.x * blockDim.x + threadIdx.x;   // 65536 total
    int j    = gid & 15;
    int lane = (gid >> 4) & 31;
    int f    = gid >> 9;            // 0..127
    int kc   = f & 15;
    int nt   = f >> 4;
    int k = kc * 32 + ((lane < 16) ? 0 : 8) + (j & 7) + ((j >= 8) ? 16 : 0);
    int n = nt * 16 + (lane & 15);
    w1p[gid] = (_Float16)W1[k * H_ + n];
}

// ---------------------------------------------------------------------------
// Kernel 1: gate.  importance[b,t] = gelu(x@W1+b1)@W2 + b2   (temperature=1)
// 256 threads = 8 waves; wave handles 16 tokens x full 128 hidden.
// ---------------------------------------------------------------------------
__global__ __launch_bounds__(256)
void gate_kernel(const float* __restrict__ x,
                 const _Float16* __restrict__ w1p,
                 const float* __restrict__ b1,
                 const float* __restrict__ W2,
                 const float* __restrict__ b2,
                 float* __restrict__ imp) {
    const int lane    = threadIdx.x & 31;
    const int wave    = threadIdx.x >> 5;
    const int tokBase = blockIdx.x * 128 + wave * 16;
    const int row     = lane & 15;
    const float* xrow = x + (size_t)(tokBase + row) * C_;
    const int off     = (lane & 16) ? 8 : 0;   // K sub-offset per half-wave

    v8f acc[8] = {};   // 8 hidden tiles of 16x16 f32

    for (int kc = 0; kc < 16; ++kc) {
        const int kbase = kc * 32;
        // A fragment per CDNA5 16-bit 16x32 A layout:
        // lanes<16: halves 0..7 -> K kbase+0..7, halves 8..15 -> kbase+16..23
        // lanes>=16: runs shifted by +8.
        const float4* p0 = (const float4*)(xrow + kbase + off);
        const float4* p1 = (const float4*)(xrow + kbase + off + 16);
        float4 f0 = p0[0], f1 = p0[1], f2 = p1[0], f3 = p1[1];
        v16h a;
        a[0]=(_Float16)f0.x; a[1]=(_Float16)f0.y; a[2]=(_Float16)f0.z; a[3]=(_Float16)f0.w;
        a[4]=(_Float16)f1.x; a[5]=(_Float16)f1.y; a[6]=(_Float16)f1.z; a[7]=(_Float16)f1.w;
        a[8]=(_Float16)f2.x; a[9]=(_Float16)f2.y; a[10]=(_Float16)f2.z; a[11]=(_Float16)f2.w;
        a[12]=(_Float16)f3.x; a[13]=(_Float16)f3.y; a[14]=(_Float16)f3.z; a[15]=(_Float16)f3.w;

        #pragma unroll
        for (int nt = 0; nt < 8; ++nt) {
            v16h bfrag = *(const v16h*)(w1p + ((((nt * 16 + kc) * 32) + lane) << 4));
            acc[nt] = __builtin_amdgcn_wmma_f32_16x16x32_f16(
                false, a, false, bfrag, (short)0, acc[nt], false, false);
        }
    }

    // h -> gelu -> dot with W2.  C/D layout: lane holds column (lane&15),
    // VGPR r holds row r + 8*(lane>=16).
    float tok[8] = {0.f,0.f,0.f,0.f,0.f,0.f,0.f,0.f};
    #pragma unroll
    for (int nt = 0; nt < 8; ++nt) {
        int n = nt * 16 + (lane & 15);
        float b1v = b1[n];
        float w2v = W2[n];
        #pragma unroll
        for (int r = 0; r < 8; ++r) {
            float h = acc[nt][r] + b1v;
            float g = 0.5f * h * (1.0f + erff(h * 0.70710678118654752440f));
            tok[r] += g * w2v;
        }
    }
    // reduce across the 16 lanes holding the same token row (halves disjoint)
    #pragma unroll
    for (int r = 0; r < 8; ++r)
        #pragma unroll
        for (int m = 1; m < 16; m <<= 1)
            tok[r] += __shfl_xor(tok[r], m, 32);

    float b2v = b2[0];
    if (lane == 0) {
        #pragma unroll
        for (int r = 0; r < 8; ++r) imp[tokBase + r] = tok[r] + b2v;
    }
    if (lane == 16) {
        #pragma unroll
        for (int r = 0; r < 8; ++r) imp[tokBase + 8 + r] = tok[r] + b2v;
    }
}

// ---------------------------------------------------------------------------
// Kernel 2: per-(b,region) softmax over member tokens + weighted gather of x.
// One block per (b, region id n0+1).  Deterministic: fixed-order reductions,
// order-preserving compaction, fixed accumulation order.
// ---------------------------------------------------------------------------
__global__ __launch_bounds__(256)
void region_gather_kernel(const float* __restrict__ x,
                          const float* __restrict__ imp,
                          const int* __restrict__ regions,
                          float* __restrict__ out) {
    __shared__ float          red[256];
    __shared__ int            s_cnt[257];
    __shared__ unsigned short s_idx[T_];   // 16 KB
    __shared__ float          s_wt[T_];    // 32 KB

    const int tid = threadIdx.x;
    const int b   = blockIdx.x / N_;
    const int n0  = blockIdx.x % N_;
    const int rid = n0 + 1;                // region ids 1..N (0 = excluded)
    const int*   reg = regions + (size_t)b * T_;
    const float* ip  = imp     + (size_t)b * T_;

    // ---- 1. region max (order independent) ----
    float lmax = -INFINITY;
    for (int t = tid; t < T_; t += 256)
        if (reg[t] == rid) lmax = fmaxf(lmax, ip[t]);
    red[tid] = lmax;
    __syncthreads();
    for (int s = 128; s > 0; s >>= 1) {
        if (tid < s) red[tid] = fmaxf(red[tid], red[tid + s]);
        __syncthreads();
    }
    const float gmax = red[0];
    __syncthreads();

    // ---- 2. per-thread contiguous chunk: count + local exp-sum ----
    const int t0 = tid * 32;               // 256 * 32 == 8192
    int cnt = 0; float lsum = 0.f;
    for (int t = t0; t < t0 + 32; ++t)
        if (reg[t] == rid) { ++cnt; lsum += expf(ip[t] - gmax); }
    s_cnt[tid] = cnt;
    red[tid]   = lsum;
    __syncthreads();
    // deterministic tree sum for Z
    for (int s = 128; s > 0; s >>= 1) {
        if (tid < s) red[tid] += red[tid + s];
        __syncthreads();
    }
    const float Z = red[0];
    __syncthreads();
    // serial exclusive prefix scan (256 iters, negligible)
    if (tid == 0) {
        int run = 0;
        for (int i = 0; i < 256; ++i) { int c = s_cnt[i]; s_cnt[i] = run; run += c; }
        s_cnt[256] = run;
    }
    __syncthreads();
    const int K = s_cnt[256];

    // ---- 3. order-preserving compaction into LDS ----
    int pos = s_cnt[tid];
    for (int t = t0; t < t0 + 32; ++t) {
        if (reg[t] == rid) {
            s_idx[pos] = (unsigned short)t;
            s_wt[pos]  = expf(ip[t] - gmax);
            ++pos;
        }
    }
    __syncthreads();

    // ---- 4. gather: thread owns channels (2*tid, 2*tid+1); fixed order j ----
    const float invZ = 1.0f / Z;           // empty region -> inf*0 = NaN (ref NaN too)
    const float2* xb = (const float2*)(x + (size_t)b * T_ * C_);
    float2 acc; acc.x = 0.f; acc.y = 0.f;
    for (int j = 0; j < K; ++j) {
        const float w = s_wt[j];                       // LDS broadcast
        const int   t = s_idx[j];
        const float2 v = xb[(size_t)t * (C_ / 2) + tid];
        acc.x += w * v.x;
        acc.y += w * v.y;
    }
    acc.x *= invZ; acc.y *= invZ;
    float2* op = (float2*)(out + ((size_t)b * N_ + n0) * C_);
    op[tid] = acc;
}

// ---------------------------------------------------------------------------
extern "C" void kernel_launch(void* const* d_in, const int* in_sizes, int n_in,
                              void* d_out, int out_size, void* d_ws, size_t ws_size,
                              hipStream_t stream) {
    const float* x       = (const float*)d_in[0];
    const float* W1      = (const float*)d_in[1];
    const float* b1      = (const float*)d_in[2];
    const float* W2      = (const float*)d_in[3];
    const float* b2      = (const float*)d_in[4];
    const int*   regions = (const int*)  d_in[5];
    // d_in[6] = max_n (128), compile-time constant here
    float* out = (float*)d_out;

    // workspace: [0,128KB) packed f16 W1 fragments, [128KB, 128KB+512KB) importance
    _Float16* w1p = (_Float16*)d_ws;
    float*    imp = (float*)((char*)d_ws + 128 * 1024);

    prepack_w1_kernel<<<256, 256, 0, stream>>>(W1, w1p);                       // 65536 elems
    gate_kernel<<<(B_ * T_) / 128, 256, 0, stream>>>(x, w1p, b1, W2, b2, imp); // 1024 blocks
    region_gather_kernel<<<B_ * N_, 256, 0, stream>>>(x, imp, regions, out);   // 2048 blocks
}
